// HGT_46205258170455
// MI455X (gfx1250) — matrix operand order repaired and checked by reading, not compiled, not run on hardware
//
#include <hip/hip_runtime.h>

// HGT forward for MI455X / gfx1250.
//  - Dense projections: bf16 WMMA (v_wmma_f32_16x16x32_bf16, fp32 accum).
//    Weights pre-packed per call into WMMA B-fragment layout (one 32B load
//    per fragment); each wave computes a 16x32 tile so every A-fragment
//    (4x b128 load + 8x v_cvt_pk_bf16_f32) feeds TWO wmma ops.
//  - Edge softmax/aggregation: L2-resident gathers + float atomics (node
//    feature tensors ~8.4MB each << 192MB L2).
// Workspace: ~126 MB fp32 + ~0.7 MB packed bf16 weights.

#define HIDF 128
#define DHH  32

typedef __attribute__((ext_vector_type(16))) __bf16 v16bf;
typedef __attribute__((ext_vector_type(8)))  float  v8f;

__device__ __forceinline__ float gelu_f(float x) {
  return 0.5f * x * (1.0f + erff(x * 0.7071067811865475f));
}

__device__ __forceinline__ void atomic_max_f32(float* a, float v) {
  if (v >= 0.0f) atomicMax((int*)a, __float_as_int(v));
  else           atomicMin((unsigned int*)a, (unsigned int)__float_as_int(v));
}

// 16x32 bf16 A-fragment (lane<16: K 0..7/16..23, lane>=16: +8):
// four float4 loads + packed converts. ap = A[row] + kk + hi*8.
__device__ __forceinline__ v16bf load_a_frag(const float* ap) {
  float4 a0 = *(const float4*)(ap);
  float4 a1 = *(const float4*)(ap + 4);
  float4 a2 = *(const float4*)(ap + 16);
  float4 a3 = *(const float4*)(ap + 20);
  v16bf a;
  a[0]=(__bf16)a0.x;  a[1]=(__bf16)a0.y;  a[2]=(__bf16)a0.z;  a[3]=(__bf16)a0.w;
  a[4]=(__bf16)a1.x;  a[5]=(__bf16)a1.y;  a[6]=(__bf16)a1.z;  a[7]=(__bf16)a1.w;
  a[8]=(__bf16)a2.x;  a[9]=(__bf16)a2.y;  a[10]=(__bf16)a2.z; a[11]=(__bf16)a2.w;
  a[12]=(__bf16)a3.x; a[13]=(__bf16)a3.y; a[14]=(__bf16)a3.z; a[15]=(__bf16)a3.w;
  return a;
}

// ---------------------------------------------------------------------------
// Weight pre-pack: fp32 [nb][K x M] (row stride ldw, batch stride bstride)
// -> bf16 fragments out[(((b*(K/32)+kt)*(M/16)+ct)*32 + lane)*16 + i]
// where k = kt*32 + (lane>=16 ? 8 : 0) + i + (i>=8 ? 8 : 0), col = ct*16 + (lane&15).
// ---------------------------------------------------------------------------
__global__ void k_pack_w(const float* __restrict__ W, __bf16* __restrict__ out,
                         int K, int M, int ldw, int nb, size_t bstride) {
  int t = blockIdx.x * blockDim.x + threadIdx.x;
  if (t >= nb * K * M) return;
  int i    = t & 15;
  int lane = (t >> 4) & 31;
  int rest = t >> 9;
  int nct  = M >> 4, nkt = K >> 5;
  int ct = rest % nct; rest /= nct;
  int kt = rest % nkt; rest /= nkt;
  int b  = rest;
  int k   = (kt << 5) + ((lane >> 4) << 3) + i + ((i >> 3) << 3);
  int col = (ct << 4) + (lane & 15);
  out[t] = (__bf16)W[(size_t)b * bstride + (size_t)k * ldw + col];
}

// ---------------------------------------------------------------------------
// WMMA GEMM: C = epi(A[N,K] @ W[K,M] + bias). 128 thr = 4 wave32; each wave
// computes a 16x32 tile (2 wmma per A-fragment). grid = (N/16, M/128).
// Branch-free inner loop -> EXEC stays all-ones around WMMA.
// ---------------------------------------------------------------------------
template<bool RELU, bool BLEND>
__global__ __launch_bounds__(128) void k_gemm(
    const float* __restrict__ A, int lda,
    const __bf16* __restrict__ Wp, int nct,       // packed weights, M/16
    const float* __restrict__ bias,
    float* C, int ldc,
    float* C2, int ldc2,                          // concat buffer (BLEND)
    const float* X, int ldx,                      // may alias C (BLEND)
    const float* __restrict__ alpha_ptr,          // sigmoid(skip) (BLEND)
    int K)
{
  const int wave = threadIdx.x >> 5;
  const int lane = threadIdx.x & 31;
  const int r0  = blockIdx.x << 4;
  const int ct0 = (blockIdx.y << 3) + (wave << 1); // two 16-col tiles per wave
  const int hi  = lane >> 4;
  const int arow = r0 + (lane & 15);

  v8f acc0 = {0.f,0.f,0.f,0.f,0.f,0.f,0.f,0.f};
  v8f acc1 = {0.f,0.f,0.f,0.f,0.f,0.f,0.f,0.f};
  const __bf16* bl = Wp + (lane << 4);
  for (int kk = 0; kk < K; kk += 32) {
    v16bf a = load_a_frag(A + (size_t)arow * lda + kk + (hi << 3));
    const __bf16* bk = bl + ((size_t)((kk >> 5) * nct + ct0) << 9);
    v16bf b0 = *(const v16bf*)(bk);
    v16bf b1 = *(const v16bf*)(bk + 512);
    acc0 = __builtin_amdgcn_wmma_f32_16x16x32_bf16(false, a, false, b0,
                                                   (short)0, acc0, false, false);
    acc1 = __builtin_amdgcn_wmma_f32_16x16x32_bf16(false, a, false, b1,
                                                   (short)0, acc1, false, false);
  }

  const int col0 = (ct0 << 4) + (lane & 15);
  const int col1 = col0 + 16;
  const float bv0 = bias[col0];
  const float bv1 = bias[col1];
  float alpha = 1.0f, beta = 0.0f;
  if (BLEND) { alpha = *alpha_ptr; beta = 1.0f - alpha; }
  const int rbase = r0 + (hi << 3);
#pragma unroll
  for (int v = 0; v < 8; ++v) {
    int row = rbase + v;
    float v0 = acc0[v] + bv0;
    float v1 = acc1[v] + bv1;
    if (RELU) { v0 = fmaxf(v0, 0.0f); v1 = fmaxf(v1, 0.0f); }
    if (BLEND) {
      v0 = alpha * v0 + beta * X[(size_t)row * ldx + col0];
      v1 = alpha * v1 + beta * X[(size_t)row * ldx + col1];
      C2[(size_t)row * ldc2 + col0] = v0;
      C2[(size_t)row * ldc2 + col1] = v1;
    }
    C[(size_t)row * ldc + col0] = v0;
    C[(size_t)row * ldc + col1] = v1;
  }
}

// ---------------------------------------------------------------------------
// Per-head relation transforms: ke = k @ a_rel[h], ve = v @ m_rel[h].
// 128 thr = 4 waves; wave = head; each wave does the full 16x32 head output
// for BOTH ke and ve: 2 A-fragments, 4 B-fragments, 4 wmma. K=32.
// Packed rel weights: batch=4 heads of K=32,M=32 (nkt=1, nct=2).
// ---------------------------------------------------------------------------
__global__ __launch_bounds__(128) void k_rel(
    const float* __restrict__ kin, const float* __restrict__ vin,
    const __bf16* __restrict__ pa, const __bf16* __restrict__ pm,
    float* __restrict__ ke, float* __restrict__ ve)
{
  const int head = threadIdx.x >> 5;
  const int lane = threadIdx.x & 31;
  const int r0 = blockIdx.x << 4;
  const int hi = lane >> 4;
  const int arow = r0 + (lane & 15);

  const size_t aoff = (size_t)arow * HIDF + (head << 5) + (hi << 3);
  v16bf ak = load_a_frag(kin + aoff);
  v16bf av = load_a_frag(vin + aoff);
  const size_t foff = ((size_t)(head << 1) << 9) + (lane << 4);
  v16bf bk0 = *(const v16bf*)(pa + foff);
  v16bf bk1 = *(const v16bf*)(pa + foff + 512);
  v16bf bm0 = *(const v16bf*)(pm + foff);
  v16bf bm1 = *(const v16bf*)(pm + foff + 512);

  v8f z = {0.f,0.f,0.f,0.f,0.f,0.f,0.f,0.f};
  v8f ok0 = __builtin_amdgcn_wmma_f32_16x16x32_bf16(false, ak, false, bk0, (short)0, z, false, false);
  v8f ok1 = __builtin_amdgcn_wmma_f32_16x16x32_bf16(false, ak, false, bk1, (short)0, z, false, false);
  v8f ov0 = __builtin_amdgcn_wmma_f32_16x16x32_bf16(false, av, false, bm0, (short)0, z, false, false);
  v8f ov1 = __builtin_amdgcn_wmma_f32_16x16x32_bf16(false, av, false, bm1, (short)0, z, false, false);

  const int col0 = (head << 5) + (lane & 15);
  const int col1 = col0 + 16;
  const int rbase = r0 + (hi << 3);
#pragma unroll
  for (int v = 0; v < 8; ++v) {
    int row = rbase + v;
    ke[(size_t)row * HIDF + col0] = ok0[v];
    ke[(size_t)row * HIDF + col1] = ok1[v];
    ve[(size_t)row * HIDF + col0] = ov0[v];
    ve[(size_t)row * HIDF + col1] = ov1[v];
  }
}

// ---------------------------------------------------------------------------
// Elementwise + edge pipeline
// ---------------------------------------------------------------------------
__global__ void k_gelu(const float* __restrict__ in, float* __restrict__ out, int n) {
  int t = blockIdx.x * blockDim.x + threadIdx.x;
  if (t < n) out[t] = gelu_f(in[t]);
}

__global__ void k_alpha(const float* s0, const float* s1,
                        const float* s2, const float* s3, float* out) {
  int t = threadIdx.x;
  if (t < 4) {
    const float* p = (t == 0) ? s0 : (t == 1) ? s1 : (t == 2) ? s2 : s3;
    out[t] = 1.0f / (1.0f + __expf(-*p));
  }
}

__global__ void k_init_prop(float* mx, float* sm, float* agg, int ndst) {
  int t = blockIdx.x * blockDim.x + threadIdx.x;
  if (t < ndst * 4)    { mx[t] = -__builtin_inff(); sm[t] = 0.0f; }
  if (t < ndst * HIDF) { agg[t] = 0.0f; }
}

// wave per edge: score[e,h] = p_rel[h]/sqrt(DH) * <q[dst,h,:], ke[src,h,:]>
__global__ void k_edge_score(const int* __restrict__ ei, int E,
                             const float* __restrict__ q,
                             const float* __restrict__ ke,
                             const float* __restrict__ p_rel,
                             float* __restrict__ ex, float* mx) {
  int e = blockIdx.x * (blockDim.x >> 5) + (threadIdx.x >> 5);
  if (e >= E) return;
  int lane = threadIdx.x & 31;
  int src = ei[e], dst = ei[E + e];
  float4 qv = *(const float4*)(q  + (size_t)dst * HIDF + lane * 4);
  float4 kv = *(const float4*)(ke + (size_t)src * HIDF + lane * 4);
  float part = qv.x*kv.x + qv.y*kv.y + qv.z*kv.z + qv.w*kv.w;
  part += __shfl_xor(part, 1);
  part += __shfl_xor(part, 2);
  part += __shfl_xor(part, 4);               // reduce 8-lane head groups
  float sval = 0.0f;
  if ((lane & 7) == 0)
    sval = part * p_rel[lane >> 3] * 0.17677669529663688f;  // 1/sqrt(32)
  float s0 = __shfl(sval, 0), s1 = __shfl(sval, 8);
  float s2 = __shfl(sval, 16), s3 = __shfl(sval, 24);
  if (lane == 0) *(float4*)(ex + (size_t)e * 4) = make_float4(s0, s1, s2, s3);
  if ((lane & 7) == 0)
    atomic_max_f32(mx + (size_t)dst * 4 + (lane >> 3), sval);
}

// thread per edge: ex = exp(score - mx[dst]); sm[dst] += ex  (vectorized x4)
__global__ void k_edge_exp(const int* __restrict__ ei, int E,
                           float* __restrict__ ex,
                           const float* __restrict__ mx, float* sm) {
  int e = blockIdx.x * blockDim.x + threadIdx.x;
  if (e >= E) return;
  int dst = ei[E + e];
  float4 s = *(const float4*)(ex + (size_t)e * 4);
  float4 m = *(const float4*)(mx + (size_t)dst * 4);
  float4 v;
  v.x = __expf(s.x - m.x); v.y = __expf(s.y - m.y);   // args <= 0: native exp
  v.z = __expf(s.z - m.z); v.w = __expf(s.w - m.w);
  *(float4*)(ex + (size_t)e * 4) = v;
  float* sp = sm + (size_t)dst * 4;
  atomicAdd(sp + 0, v.x); atomicAdd(sp + 1, v.y);
  atomicAdd(sp + 2, v.z); atomicAdd(sp + 3, v.w);
}

// wave per edge: agg[dst] += att[e,h] * ve[src]  (coalesced 128-float scatter)
__global__ void k_edge_agg(const int* __restrict__ ei, int E,
                           const float* __restrict__ ex,
                           const float* __restrict__ sm,
                           const float* __restrict__ ve, float* agg) {
  int e = blockIdx.x * (blockDim.x >> 5) + (threadIdx.x >> 5);
  if (e >= E) return;
  int lane = threadIdx.x & 31;
  int src = ei[e], dst = ei[E + e];
  float4 exv = *(const float4*)(ex + (size_t)e * 4);
  float4 smv = *(const float4*)(sm + (size_t)dst * 4);
  float att[4] = { exv.x / (smv.x + 1e-16f), exv.y / (smv.y + 1e-16f),
                   exv.z / (smv.z + 1e-16f), exv.w / (smv.w + 1e-16f) };
#pragma unroll
  for (int h = 0; h < 4; ++h) {
    int comp = (h << 5) + lane;
    atomicAdd(agg + (size_t)dst * HIDF + comp,
              ve[(size_t)src * HIDF + comp] * att[h]);
  }
}

// wave per pair: y[p] = <Em[m,:256], Ed[d,:256]>
__global__ void k_sddmm(const int* __restrict__ ei, int P,
                        const float* __restrict__ Em,
                        const float* __restrict__ Ed, float* y) {
  int p = blockIdx.x * (blockDim.x >> 5) + (threadIdx.x >> 5);
  if (p >= P) return;
  int lane = threadIdx.x & 31;
  int m = ei[p], d = ei[P + p];
  const float4* a = (const float4*)(Em + (size_t)m * 256);
  const float4* b = (const float4*)(Ed + (size_t)d * 256);
  float s = 0.0f;
#pragma unroll
  for (int i = 0; i < 2; ++i) {
    float4 av = a[lane + 32 * i], bv = b[lane + 32 * i];
    s += av.x*bv.x + av.y*bv.y + av.z*bv.z + av.w*bv.w;
  }
  s += __shfl_xor(s, 16); s += __shfl_xor(s, 8); s += __shfl_xor(s, 4);
  s += __shfl_xor(s, 2);  s += __shfl_xor(s, 1);
  if (lane == 0) y[p] = s;
}

// ---------------------------------------------------------------------------
// Host orchestration
// ---------------------------------------------------------------------------
struct NodeP { const float *kW,*kb,*qW,*qb,*vW,*vb,*aW,*ab,*skip; };
struct RelP  { const float *a_rel,*m_rel,*p_rel; };
struct LayerP { NodeP n1, n2; RelP e12, e21; };
struct NodePk { __bf16 *k,*q,*v,*a; };
struct LayerPk { NodePk n1, n2; __bf16 *ra12,*rm12,*ra21,*rm21; };

extern "C" void kernel_launch(void* const* d_in, const int* in_sizes, int n_in,
                              void* d_out, int out_size, void* d_ws, size_t ws_size,
                              hipStream_t stream) {
  (void)n_in; (void)out_size; (void)ws_size;
  const float* x1    = (const float*)d_in[0];
  const float* x2    = (const float*)d_in[1];
  const int*   e12   = (const int*)d_in[2];
  const int*   e21   = (const int*)d_in[3];
  const int*   epair = (const int*)d_in[4];
  const int N1 = in_sizes[0] / 256;
  const int N2 = in_sizes[1] / 256;
  const int E  = in_sizes[2] / 2;
  const int P  = in_sizes[4] / 2;

  auto F = [&](int i) { return (const float*)d_in[i]; };
  LayerP L[2];
  const float *linW1, *linb1, *linW2, *linb2;
  if (in_sizes[5] == 4096) {
    // jax tree_leaves order (sorted dict keys at every level)
    for (int l = 0; l < 2; ++l) {
      int b = 5 + 24 * l;
      L[l].e12 = { F(b+0),  F(b+1),  F(b+2)  };
      L[l].e21 = { F(b+3),  F(b+4),  F(b+5)  };
      L[l].n1  = { F(b+8),  F(b+9),  F(b+10), F(b+11), F(b+13), F(b+14),
                   F(b+6),  F(b+7),  F(b+12) };
      L[l].n2  = { F(b+17), F(b+18), F(b+19), F(b+20), F(b+22), F(b+23),
                   F(b+15), F(b+16), F(b+21) };
    }
    linW1 = F(53); linb1 = F(54); linW2 = F(55); linb2 = F(56);
  } else {
    // insertion-order flattening: lin_n1, lin_n2, layers (n1:k,q,v,a,skip; n2; e12; e21)
    linW1 = F(5); linb1 = F(6); linW2 = F(7); linb2 = F(8);
    for (int l = 0; l < 2; ++l) {
      int b = 9 + 24 * l;
      L[l].n1  = { F(b+0),  F(b+1),  F(b+2),  F(b+3),  F(b+4),  F(b+5),
                   F(b+6),  F(b+7),  F(b+8)  };
      L[l].n2  = { F(b+9),  F(b+10), F(b+11), F(b+12), F(b+13), F(b+14),
                   F(b+15), F(b+16), F(b+17) };
      L[l].e12 = { F(b+18), F(b+19), F(b+20) };
      L[l].e21 = { F(b+21), F(b+22), F(b+23) };
    }
  }

  // ---- workspace carve: fp32 region ----
  float* w = (float*)d_ws;
  auto take = [&](size_t n) { float* p = w; w += n; return p; };
  const int NSmax = (N1 > N2) ? N1 : N2;
  float* h1   = take((size_t)N1 * HIDF);
  float* h2   = take((size_t)N2 * HIDF);
  float* Em   = take((size_t)N1 * 2 * HIDF);
  float* Ed   = take((size_t)N2 * 2 * HIDF);
  float* q1   = take((size_t)N1 * HIDF);
  float* q2   = take((size_t)N2 * HIDF);
  float* tmpk = take((size_t)NSmax * HIDF);
  float* tmpv = take((size_t)NSmax * HIDF);
  float* keb  = take((size_t)NSmax * HIDF);
  float* veb  = take((size_t)NSmax * HIDF);
  float* agg1 = take((size_t)N1 * HIDF);
  float* agg2 = take((size_t)N2 * HIDF);
  float* exb  = take((size_t)E * 4);
  float* mx   = take((size_t)NSmax * 4);
  float* sm   = take((size_t)NSmax * 4);
  float* alphas = take(16);            // 4 used, padded for alignment

  // ---- bf16 packed-weight region (32B aligned: all chunks are 1KB mult) ----
  __bf16* bw = (__bf16*)w;
  auto takeb = [&](size_t n) { __bf16* p = bw; bw += n; return p; };
  __bf16* plin1 = takeb(256 * 128);
  __bf16* plin2 = takeb(256 * 128);
  LayerPk Pk[2];
  for (int l = 0; l < 2; ++l) {
    Pk[l].n1 = { takeb(16384), takeb(16384), takeb(16384), takeb(16384) };
    Pk[l].n2 = { takeb(16384), takeb(16384), takeb(16384), takeb(16384) };
    Pk[l].ra12 = takeb(4096); Pk[l].rm12 = takeb(4096);
    Pk[l].ra21 = takeb(4096); Pk[l].rm21 = takeb(4096);
  }

  // ---- pack all weights ----
  auto pack = [&](const float* W, __bf16* out, int K, int M, int ldw,
                  int nb, size_t bstride) {
    int total = nb * K * M;
    k_pack_w<<<dim3((total + 255) / 256), dim3(256), 0, stream>>>(
        W, out, K, M, ldw, nb, bstride);
  };
  pack(linW1, plin1, 256, HIDF, HIDF, 1, 0);
  pack(linW2, plin2, 256, HIDF, HIDF, 1, 0);
  for (int l = 0; l < 2; ++l) {
    pack(L[l].n1.kW, Pk[l].n1.k, HIDF, HIDF, HIDF, 1, 0);
    pack(L[l].n1.qW, Pk[l].n1.q, HIDF, HIDF, HIDF, 1, 0);
    pack(L[l].n1.vW, Pk[l].n1.v, HIDF, HIDF, HIDF, 1, 0);
    pack(L[l].n1.aW, Pk[l].n1.a, HIDF, HIDF, HIDF, 1, 0);
    pack(L[l].n2.kW, Pk[l].n2.k, HIDF, HIDF, HIDF, 1, 0);
    pack(L[l].n2.qW, Pk[l].n2.q, HIDF, HIDF, HIDF, 1, 0);
    pack(L[l].n2.vW, Pk[l].n2.v, HIDF, HIDF, HIDF, 1, 0);
    pack(L[l].n2.aW, Pk[l].n2.a, HIDF, HIDF, HIDF, 1, 0);
    pack(L[l].e12.a_rel, Pk[l].ra12, DHH, DHH, DHH, 4, DHH * DHH);
    pack(L[l].e12.m_rel, Pk[l].rm12, DHH, DHH, DHH, 4, DHH * DHH);
    pack(L[l].e21.a_rel, Pk[l].ra21, DHH, DHH, DHH, 4, DHH * DHH);
    pack(L[l].e21.m_rel, Pk[l].rm21, DHH, DHH, DHH, 4, DHH * DHH);
  }
  k_alpha<<<dim3(1), dim3(4), 0, stream>>>(
      L[0].n1.skip, L[0].n2.skip, L[1].n1.skip, L[1].n2.skip, alphas);

  const dim3 gB(128);
  auto gemm_plain = [&](const float* A, int lda, const __bf16* Wp,
                        const float* bias, float* C, int N, int K, bool relu) {
    if (relu)
      k_gemm<true, false><<<dim3(N/16, 1), gB, 0, stream>>>(
          A, lda, Wp, HIDF/16, bias, C, HIDF, nullptr, 0, nullptr, 0, nullptr, K);
    else
      k_gemm<false, false><<<dim3(N/16, 1), gB, 0, stream>>>(
          A, lda, Wp, HIDF/16, bias, C, HIDF, nullptr, 0, nullptr, 0, nullptr, K);
  };

  // input projections + ReLU
  gemm_plain(x1, 256, plin1, linb1, h1, N1, 256, true);
  gemm_plain(x2, 256, plin2, linb2, h2, N2, 256, true);

  auto run_prop = [&](const float* hsrc, int nsrc, const NodeP& np,
                      const NodePk& npk, const RelP& rp,
                      const __bf16* pra, const __bf16* prm,
                      const int* ei, const float* qdst, int ndst, float* agg) {
    gemm_plain(hsrc, HIDF, npk.k, np.kb, tmpk, nsrc, HIDF, false);
    gemm_plain(hsrc, HIDF, npk.v, np.vb, tmpv, nsrc, HIDF, false);
    k_rel<<<dim3(nsrc/16), gB, 0, stream>>>(tmpk, tmpv, pra, prm, keb, veb);
    k_init_prop<<<dim3((ndst*HIDF + 255)/256), dim3(256), 0, stream>>>(mx, sm, agg, ndst);
    k_edge_score<<<dim3((E + 7)/8), dim3(256), 0, stream>>>(ei, E, qdst, keb,
        rp.p_rel, exb, mx);
    k_edge_exp<<<dim3((E + 255)/256), dim3(256), 0, stream>>>(ei, E, exb, mx, sm);
    k_edge_agg<<<dim3((E + 7)/8), dim3(256), 0, stream>>>(ei, E, exb, sm, veb, agg);
  };

  for (int l = 0; l < 2; ++l) {
    const LayerP& lp = L[l];
    const LayerPk& pk = Pk[l];
    // q projections for both node types
    gemm_plain(h1, HIDF, pk.n1.q, lp.n1.qb, q1, N1, HIDF, false);
    gemm_plain(h2, HIDF, pk.n2.q, lp.n2.qb, q2, N2, HIDF, false);
    // e12: src n1 -> dst n2 (agg2); e21: src n2 -> dst n1 (agg1)
    run_prop(h1, N1, lp.n1, pk.n1, lp.e12, pk.ra12, pk.rm12, e12, q2, N2, agg2);
    run_prop(h2, N2, lp.n2, pk.n2, lp.e21, pk.ra21, pk.rm21, e21, q1, N1, agg1);
    // update: h = a*lin(gelu(agg)) + (1-a)*h ; dual-store into concat buffer
    k_gelu<<<dim3((N1*HIDF + 255)/256), dim3(256), 0, stream>>>(agg1, tmpk, N1*HIDF);
    k_gelu<<<dim3((N2*HIDF + 255)/256), dim3(256), 0, stream>>>(agg2, tmpv, N2*HIDF);
    k_gemm<false, true><<<dim3(N1/16, 1), gB, 0, stream>>>(
        tmpk, HIDF, pk.n1.a, HIDF/16, lp.n1.ab, h1, HIDF,
        Em + l*HIDF, 2*HIDF, h1, HIDF, alphas + 2*l + 0, HIDF);
    k_gemm<false, true><<<dim3(N2/16, 1), gB, 0, stream>>>(
        tmpv, HIDF, pk.n2.a, HIDF/16, lp.n2.ab, h2, HIDF,
        Ed + l*HIDF, 2*HIDF, h2, HIDF, alphas + 2*l + 1, HIDF);
  }

  k_sddmm<<<dim3((P + 7)/8), dim3(256), 0, stream>>>(epair, P, Em, Ed, (float*)d_out);
}
